// BeamSearchGenerator_42056319762607
// MI455X (gfx1250) — matrix operand order repaired and checked by reading, not compiled, not run on hardware
//
#include <hip/hip_runtime.h>
#include <hip/hip_bf16.h>

#define BOS 1
#define EOS 2
#define MAX_LEN 16
#define BEAM 4
#define VOC 32000
#define NB 16          // batch
#define SRC_S 128
#define DIM 512
#define ROWS 64        // NB*BEAM
#define NEG_BIG 1.0e9f

typedef __attribute__((ext_vector_type(16))) __bf16        v16bf;
typedef __attribute__((ext_vector_type(8)))  float         v8f;
typedef __attribute__((ext_vector_type(8)))  unsigned int  v8u;
typedef __attribute__((ext_vector_type(4)))  unsigned int  v4u;

__device__ __forceinline__ unsigned short f2bf(float x) {
    unsigned int u = __builtin_bit_cast(unsigned int, x);
    unsigned int r = u + 0x7FFFu + ((u >> 16) & 1u);   // round-to-nearest-even
    return (unsigned short)(r >> 16);
}

// ---------------- init state ----------------
__global__ void init_state_k(int* alive_seq, float* alive_lp, int* fin_seq,
                             float* fin_sc, int* fin_fl, int* batch_fin) {
    int tid = threadIdx.x;
    for (int i = tid; i < NB * BEAM * MAX_LEN; i += 256) {
        alive_seq[i] = ((i & (MAX_LEN - 1)) == 0) ? BOS : 0;
        fin_seq[i] = 0;
    }
    for (int i = tid; i < NB * BEAM; i += 256) {
        alive_lp[i] = ((i & (BEAM - 1)) == 0) ? 0.0f : -__builtin_inff();
        fin_sc[i] = -NEG_BIG;
        fin_fl[i] = 0;
    }
    for (int i = tid; i < NB; i += 256) batch_fin[i] = 0;
}

// ---------------- encoder pool: pooled[b][d] ----------------
__global__ __launch_bounds__(256) void pool_k(const int* __restrict__ src,
                                              const float* __restrict__ Es,
                                              float* __restrict__ pooled) {
    __shared__ int toks[SRC_S];
    __shared__ float cnt_s;
    int b = blockIdx.x, tid = threadIdx.x;
    if (tid < SRC_S) toks[tid] = src[b * SRC_S + tid];
    __syncthreads();
    if (tid == 0) {
        int c = 0;
        for (int s = 0; s < SRC_S; s++) c += (toks[s] != 0);
        cnt_s = fmaxf((float)c, 1.0f);
    }
    __syncthreads();
    float inv = 1.0f / cnt_s;
    for (int d = tid; d < DIM; d += 256) {
        float acc = 0.0f;
        for (int s = 0; s < SRC_S; s++) {
            int tk = toks[s];
            if (tk != 0) acc += Es[(size_t)tk * DIM + d];
        }
        pooled[b * DIM + d] = acc * inv;
    }
}

// ---------------- W [D][V] f32 -> Wt [V][D] bf16 (tiled transpose) ----------------
__global__ __launch_bounds__(256) void convw_k(const float* __restrict__ W,
                                               unsigned short* __restrict__ Wt) {
    __shared__ float tile[32][33];
    int v0 = blockIdx.x * 32, d0 = blockIdx.y * 32;
    int tx = threadIdx.x, ty = threadIdx.y;   // (32, 8)
#pragma unroll
    for (int i = 0; i < 4; i++) {
        int d = d0 + ty + i * 8;
        tile[ty + i * 8][tx] = W[(size_t)d * VOC + v0 + tx];
    }
    __syncthreads();
#pragma unroll
    for (int i = 0; i < 4; i++) {
        int v = v0 + ty + i * 8;
        Wt[(size_t)v * DIM + d0 + tx] = f2bf(tile[tx][ty + i * 8]);
    }
}

// ---------------- x = bf16(E_tgt[last] + pooled) ----------------
__global__ __launch_bounds__(256) void xbuild_k(const float* __restrict__ Et,
                                                const float* __restrict__ pooled,
                                                const int* __restrict__ alive_seq,
                                                int t, unsigned short* __restrict__ xbf) {
    int row = blockIdx.x;           // 0..63
    int b = row >> 2, k = row & 3;
    int last = alive_seq[(b * BEAM + k) * MAX_LEN + t];
    for (int d = threadIdx.x; d < DIM; d += 256) {
        float v = Et[(size_t)last * DIM + d] + pooled[b * DIM + d];
        xbf[row * DIM + d] = f2bf(v);
    }
}

// ---------------- WMMA GEMM: logits[64][32000] = x @ W + b ----------------
// grid 250, block 256 (8 waves). Block tile: M=64, N=128. K chunked by 128 via LDS.
__global__ __launch_bounds__(256) void gemm_wmma_k(const unsigned short* __restrict__ xbf,
                                                   const unsigned short* __restrict__ wt,
                                                   const float* __restrict__ bias,
                                                   float* __restrict__ logits) {
    __shared__ unsigned int lds_x[ROWS * 64];   // 64 rows x 128 bf16 (16 KB)
    const unsigned int* xu = (const unsigned int*)xbf;  // row stride 256 dwords
    const unsigned int* wu = (const unsigned int*)wt;   // row stride 256 dwords

    int tid = threadIdx.x;
    int wv = tid >> 5;
    int l  = tid & 31;
    int lh = l & 15;
    int hi = l >> 4;
    int n  = blockIdx.x * 128 + wv * 16 + lh;   // global column this lane owns in B/C

    const v8f zero8 = {0.f, 0.f, 0.f, 0.f, 0.f, 0.f, 0.f, 0.f};
    v8f acc[4];
#pragma unroll
    for (int i = 0; i < 4; i++) acc[i] = zero8;

    for (int kc = 0; kc < 4; kc++) {            // K chunks of 128
        __syncthreads();
#pragma unroll
        for (int i = 0; i < 16; i++) {          // stage x chunk into LDS
            int idx = tid + i * 256;            // 4096 dwords total
            int r = idx >> 6, c = idx & 63;
            lds_x[idx] = xu[r * 256 + kc * 64 + c];
        }
        __syncthreads();
#pragma unroll
        for (int kk = 0; kk < 4; kk++) {        // 4 x (16x16x32) along K
            // B fragment (dense 32x16 bf16): lane lh = column n, hi selects K half.
            const v4u* bp = (const v4u*)(wu + (size_t)n * 256 + kc * 64 + kk * 16 + hi * 8);
            v4u b0 = bp[0];
            v4u b1 = bp[1];
            v8u bu = __builtin_shufflevector(b0, b1, 0, 1, 2, 3, 4, 5, 6, 7);
            v16bf bf = __builtin_bit_cast(v16bf, bu);
#pragma unroll
            for (int mt = 0; mt < 4; mt++) {
                // A fragment (16x32 bf16): lane lh = row, hi selects K sub-range.
                const unsigned int* arow = &lds_x[(mt * 16 + lh) * 64 + kk * 16 + hi * 8];
                v4u a0 = *(const v4u*)(arow);
                v4u a1 = *(const v4u*)(arow + 4);
                v8u au = __builtin_shufflevector(a0, a1, 0, 1, 2, 3, 4, 5, 6, 7);
                v16bf af = __builtin_bit_cast(v16bf, au);
                acc[mt] = __builtin_amdgcn_wmma_f32_16x16x32_bf16(
                    false, af, false, bf, (short)0, acc[mt], false, false);
            }
        }
    }
    float bv = bias[n];
#pragma unroll
    for (int mt = 0; mt < 4; mt++) {
        union { v8f v; float f[8]; } u;
        u.v = acc[mt];
#pragma unroll
        for (int r = 0; r < 8; r++) {
            int m = mt * 16 + hi * 8 + r;       // C/D layout: VGPR r, lane half -> M
            logits[(size_t)m * VOC + n] = u.f[r] + bv;
        }
    }
}

// ---------------- per-row logsumexp ----------------
__global__ __launch_bounds__(256) void lse_k(const float* __restrict__ logits,
                                             float* __restrict__ lse) {
    __shared__ float red[256];
    int r = blockIdx.x, tid = threadIdx.x;
    const float* base = logits + (size_t)r * VOC;
    float m = -3.4e38f;
    for (int v = tid; v < VOC; v += 256) m = fmaxf(m, base[v]);
    red[tid] = m;
    __syncthreads();
    for (int s = 128; s > 0; s >>= 1) {
        if (tid < s) red[tid] = fmaxf(red[tid], red[tid + s]);
        __syncthreads();
    }
    m = red[0];
    __syncthreads();
    float sum = 0.0f;
    for (int v = tid; v < VOC; v += 256) sum += expf(base[v] - m);
    red[tid] = sum;
    __syncthreads();
    for (int s = 128; s > 0; s >>= 1) {
        if (tid < s) red[tid] += red[tid + s];
        __syncthreads();
    }
    if (tid == 0) lse[r] = m + logf(red[0]);
}

// ---------------- top-8 per batch over BEAM*VOC scores ----------------
__global__ __launch_bounds__(256) void topk_k(const float* __restrict__ logits,
                                              const float* __restrict__ lse,
                                              const float* __restrict__ alive_lp,
                                              int t, float* __restrict__ tsc,
                                              int* __restrict__ tidx) {
    __shared__ float s_sc[2048];
    __shared__ int   s_id[2048];
    int b = blockIdx.x, tid = threadIdx.x;
    float lpen = (float)(t + 1);
    float bs[8];
    int bi[8];
#pragma unroll
    for (int i = 0; i < 8; i++) { bs[i] = -3.4e38f; bi[i] = 0x7fffffff; }
    for (int r = 0; r < BEAM; r++) {
        int row = b * BEAM + r;
        float alp = alive_lp[row];
        float ls  = lse[row];
        const float* base = logits + (size_t)row * VOC;
        for (int v = tid; v < VOC; v += 256) {
            float sc = (alp + base[v] - ls) / lpen;
            if (sc > bs[7]) {
                int j = 7;
                while (j > 0 && sc > bs[j - 1]) {
                    bs[j] = bs[j - 1]; bi[j] = bi[j - 1]; j--;
                }
                bs[j] = sc; bi[j] = r * VOC + v;
            }
        }
    }
#pragma unroll
    for (int i = 0; i < 8; i++) { s_sc[tid * 8 + i] = bs[i]; s_id[tid * 8 + i] = bi[i]; }
    __syncthreads();
    for (int stride = 128; stride >= 1; stride >>= 1) {
        if (tid < stride) {
            int A = tid, B2 = tid + stride;
            float os[8]; int oi[8];
            int ia = 0, ib = 0;
            for (int q = 0; q < 8; q++) {
                float sa = (ia < 8) ? s_sc[A * 8 + ia]  : -3.4e38f;
                float sb = (ib < 8) ? s_sc[B2 * 8 + ib] : -3.4e38f;
                int   da = (ia < 8) ? s_id[A * 8 + ia]  : 0x7fffffff;
                int   db = (ib < 8) ? s_id[B2 * 8 + ib] : 0x7fffffff;
                bool takeA = (sa > sb) || (sa == sb && da <= db);
                if (takeA) { os[q] = sa; oi[q] = da; ia++; }
                else       { os[q] = sb; oi[q] = db; ib++; }
            }
            for (int q = 0; q < 8; q++) { s_sc[A * 8 + q] = os[q]; s_id[A * 8 + q] = oi[q]; }
        }
        __syncthreads();
    }
    if (tid < 8) { tsc[b * 8 + tid] = s_sc[tid]; tidx[b * 8 + tid] = s_id[tid]; }
}

// ---------------- beam bookkeeping (1 thread per batch) ----------------
__global__ void beam_update_k(const float* __restrict__ tsc, const int* __restrict__ tidx,
                              int t, int* alive_seq, float* alive_lp, int* fin_seq,
                              float* fin_sc, int* fin_fl, int* batch_fin) {
    int b = threadIdx.x;
    if (b >= NB) return;
    float lpen = (float)(t + 1);

    int aseq[BEAM][MAX_LEN];
    for (int k = 0; k < BEAM; k++)
        for (int p = 0; p < MAX_LEN; p++)
            aseq[k][p] = alive_seq[(b * BEAM + k) * MAX_LEN + p];

    float sc8[8], lp8[8];
    int tok8[8], fin8[8], seq8[8][MAX_LEN];
    for (int i = 0; i < 8; i++) {
        sc8[i] = tsc[b * 8 + i];
        int id = tidx[b * 8 + i];
        int beam = id / VOC;
        tok8[i] = id - beam * VOC;
        lp8[i] = sc8[i] * lpen;
        fin8[i] = (tok8[i] == EOS) ? 1 : 0;
        for (int p = 0; p < MAX_LEN; p++) seq8[i][p] = aseq[beam][p];
        seq8[i][t + 1] = tok8[i];
    }

    // alive set: penalize finished candidates, keep top BEAM
    float curr[8];
    for (int i = 0; i < 8; i++) curr[i] = sc8[i] + (fin8[i] ? -NEG_BIG : 0.0f);
    int used[8] = {0, 0, 0, 0, 0, 0, 0, 0};
    float new_lp0 = 0.0f;
    for (int j = 0; j < BEAM; j++) {
        float best = -3.4e38f;
        int bidx = 0;
        for (int i = 0; i < 8; i++)
            if (!used[i] && curr[i] > best) { best = curr[i]; bidx = i; }
        used[bidx] = 1;
        for (int p = 0; p < MAX_LEN; p++)
            alive_seq[(b * BEAM + j) * MAX_LEN + p] = seq8[bidx][p];
        alive_lp[b * BEAM + j] = lp8[bidx];
        if (j == 0) new_lp0 = lp8[bidx];
    }

    // finished set: concat old fin (4) + new candidates (8), keep top BEAM
    int bfin = batch_fin[b];
    float fso[BEAM]; int ffo[BEAM], fqo[BEAM][MAX_LEN];
    for (int k = 0; k < BEAM; k++) {
        fso[k] = fin_sc[b * BEAM + k];
        ffo[k] = fin_fl[b * BEAM + k];
        for (int p = 0; p < MAX_LEN; p++) fqo[k][p] = fin_seq[(b * BEAM + k) * MAX_LEN + p];
    }
    float cs[12]; int cf[12];
    for (int k = 0; k < BEAM; k++) { cs[k] = fso[k]; cf[k] = ffo[k]; }
    for (int i = 0; i < 8; i++) {
        cs[BEAM + i] = sc8[i] + (fin8[i] ? 0.0f : -NEG_BIG) + (bfin ? -NEG_BIG : 0.0f);
        cf[BEAM + i] = fin8[i];
    }
    int cused[12];
    for (int i = 0; i < 12; i++) cused[i] = 0;
    float nfs[BEAM]; int nff[BEAM];
    for (int j = 0; j < BEAM; j++) {
        float best = -3.4e38f;
        int bidx = 0;
        for (int i = 0; i < 12; i++)
            if (!cused[i] && cs[i] > best) { best = cs[i]; bidx = i; }
        cused[bidx] = 1;
        nfs[j] = cs[bidx];
        nff[j] = cf[bidx];
        fin_sc[b * BEAM + j] = nfs[j];
        fin_fl[b * BEAM + j] = nff[j];
        if (bidx < BEAM) {
            for (int p = 0; p < MAX_LEN; p++)
                fin_seq[(b * BEAM + j) * MAX_LEN + p] = fqo[bidx][p];
        } else {
            for (int p = 0; p < MAX_LEN; p++)
                fin_seq[(b * BEAM + j) * MAX_LEN + p] = seq8[bidx - BEAM][p];
        }
    }

    // batch-finished lower-bound test
    float lb = new_lp0 / lpen;
    float lowest = 3.4e38f;
    int allf = 1;
    for (int k = 0; k < BEAM; k++) {
        lowest = fminf(lowest, nfs[k] * (nff[k] ? 1.0f : 0.0f));
        allf &= nff[k];
    }
    if (!allf) lowest += -NEG_BIG;
    if (lowest >= lb) bfin = 1;
    batch_fin[b] = bfin;
}

// ---------------- final output: fin_seq[:,0] ----------------
__global__ void finalize_k(const int* __restrict__ fin_seq, int* __restrict__ out) {
    int i = threadIdx.x;
    if (i < NB * MAX_LEN) out[i] = fin_seq[(i >> 4) * (BEAM * MAX_LEN) + (i & 15)];
}

extern "C" void kernel_launch(void* const* d_in, const int* in_sizes, int n_in,
                              void* d_out, int out_size, void* d_ws, size_t ws_size,
                              hipStream_t stream) {
    const int*   src  = (const int*)d_in[0];
    const float* Es   = (const float*)d_in[1];
    const float* Et   = (const float*)d_in[2];
    const float* W    = (const float*)d_in[3];
    const float* bias = (const float*)d_in[4];
    int* out = (int*)d_out;

    char* ws = (char*)d_ws;
    size_t o = 0;
    float* pooled = (float*)(ws + o);           o += (size_t)NB * DIM * 4;           // 32 KB
    unsigned short* xbf = (unsigned short*)(ws + o); o += (size_t)ROWS * DIM * 2;    // 64 KB
    float* logits = (float*)(ws + o);           o += (size_t)ROWS * VOC * 4;         // 8 MB
    float* lse = (float*)(ws + o);              o += ROWS * 4 + 192;
    float* tsc = (float*)(ws + o);              o += NB * 8 * 4;
    int* tidx = (int*)(ws + o);                 o += NB * 8 * 4;
    int* alive_seq = (int*)(ws + o);            o += NB * BEAM * MAX_LEN * 4;
    float* alive_lp = (float*)(ws + o);         o += NB * BEAM * 4;
    int* fin_seq = (int*)(ws + o);              o += NB * BEAM * MAX_LEN * 4;
    float* fin_sc = (float*)(ws + o);           o += NB * BEAM * 4;
    int* fin_fl = (int*)(ws + o);               o += NB * BEAM * 4;
    int* batch_fin = (int*)(ws + o);            o += 256;
    unsigned short* wt = (unsigned short*)(ws + o); o += (size_t)VOC * DIM * 2;      // 32 MB

    init_state_k<<<1, 256, 0, stream>>>(alive_seq, alive_lp, fin_seq, fin_sc, fin_fl, batch_fin);
    pool_k<<<NB, 256, 0, stream>>>(src, Es, pooled);
    convw_k<<<dim3(VOC / 32, DIM / 32), dim3(32, 8), 0, stream>>>(W, wt);

    for (int t = 0; t < MAX_LEN - 1; t++) {
        xbuild_k<<<ROWS, 256, 0, stream>>>(Et, pooled, alive_seq, t, xbf);
        gemm_wmma_k<<<VOC / 128, 256, 0, stream>>>(xbf, wt, bias, logits);
        lse_k<<<ROWS, 256, 0, stream>>>(logits, lse);
        topk_k<<<NB, 256, 0, stream>>>(logits, lse, alive_lp, t, tsc, tidx);
        beam_update_k<<<1, NB, 0, stream>>>(tsc, tidx, t, alive_seq, alive_lp,
                                            fin_seq, fin_sc, fin_fl, batch_fin);
    }
    finalize_k<<<1, 256, 0, stream>>>(fin_seq, out);
}
